// XConv_8435315769569
// MI455X (gfx1250) — compile-verified, matrix operand order checked by reading, not run on hardware
//
#include <hip/hip_runtime.h>

// ---------------------------------------------------------------------------
// X-Conv (PointCNN) pipeline for MI455X / gfx1250, wave32 + WMMA f16->f32.
// N=65536 points, K=16 neighbors, D=3, CIN=64, CD=16, CT=80, DM=2, COUT=128.
// mlp2/final WMMA operands pre-converted to f16 and loaded as 16B v8h; k_xconv builds them per element.
// ---------------------------------------------------------------------------

#define XN   65536
#define XK   16
#define XNE  (XN * XK)   // 1,048,576 edges
#define XEPS 1e-5f

typedef __attribute__((ext_vector_type(16))) _Float16 v16h;
typedef __attribute__((ext_vector_type(8)))  _Float16 v8h;
typedef __attribute__((ext_vector_type(8)))  float    v8f;

__device__ __forceinline__ v8f wmma_f16f32(v16h a, v16h b, v8f c) {
  return __builtin_amdgcn_wmma_f32_16x16x32_f16(
      false, a, false, b, (short)0, c, false, false);
}

__device__ __forceinline__ v16h mk_frag(v8h lo, v8h hi) {
  v16h r;
#pragma unroll
  for (int q = 0; q < 8; ++q) { r[q] = lo[q]; r[q + 8] = hi[q]; }
  return r;
}

__device__ __forceinline__ float elu1(float x) {
  return x > 0.f ? x : (__expf(x) - 1.f);
}

__device__ __forceinline__ float wave_sum32(float v) {
#pragma unroll
  for (int off = 16; off > 0; off >>= 1) v += __shfl_xor(v, off, 32);
  return v;
}

// ---------------------------------------------------------------------------
__global__ void k_zero(float* p, int n) {
  int i = blockIdx.x * blockDim.x + threadIdx.x;
  if (i < n) p[i] = 0.f;
}

__global__ void k_tohalf(const float* __restrict__ src, _Float16* __restrict__ dst, int n) {
  int i = blockIdx.x * blockDim.x + threadIdx.x;
  if (i < n) dst[i] = (_Float16)src[i];
}

// BN finalize: fold batch mean/var + gamma/beta into per-channel affine.
__global__ void k_finalize_bn(const float* __restrict__ sum,
                              const float* __restrict__ sqs,
                              const float* __restrict__ g,
                              const float* __restrict__ be,
                              float* __restrict__ scale,
                              float* __restrict__ shift,
                              int C, float invn) {
  int i = threadIdx.x;
  if (i < C) {
    float m = sum[i] * invn;
    float v = sqs[i] * invn - m * m;
    float s = g[i] * rsqrtf(v + XEPS);
    scale[i] = s;
    shift[i] = be[i] - m * s;
  }
}

// ---------------------------------------------------------------------------
// Stage 1: rel = pos[col]-pos[row] (stored f16 for WMMA); h1 = elu(rel@W1^T+b1).
// Stats via wave32 butterfly + one LDS/global atomic per wave.
__global__ void __launch_bounds__(256)
k_mlp1_s1(const float* __restrict__ pos, const int* __restrict__ col,
          const float* __restrict__ w1, const float* __restrict__ b1,
          _Float16* __restrict__ rel_h, float* __restrict__ h,
          float* __restrict__ gsum, float* __restrict__ gsqs) {
  __shared__ float ssum[16], ssqs[16];
  int tid = threadIdx.x;
  if (tid < 16) { ssum[tid] = 0.f; ssqs[tid] = 0.f; }
  __syncthreads();
  long e = (long)blockIdx.x * 256 + tid;
  long n = e >> 4;
  long c = col[e];
  float r0 = pos[c * 3 + 0] - pos[n * 3 + 0];
  float r1 = pos[c * 3 + 1] - pos[n * 3 + 1];
  float r2 = pos[c * 3 + 2] - pos[n * 3 + 2];
  rel_h[e * 3 + 0] = (_Float16)r0;
  rel_h[e * 3 + 1] = (_Float16)r1;
  rel_h[e * 3 + 2] = (_Float16)r2;
  float v[16];
#pragma unroll
  for (int j = 0; j < 16; ++j) {
    v[j] = elu1(r0 * w1[j * 3 + 0] + r1 * w1[j * 3 + 1] + r2 * w1[j * 3 + 2] + b1[j]);
    h[e * 16 + j] = v[j];
  }
#pragma unroll
  for (int j = 0; j < 16; ++j) {
    float s = wave_sum32(v[j]);
    float q = wave_sum32(v[j] * v[j]);
    if ((tid & 31) == 0) { atomicAdd(&ssum[j], s); atomicAdd(&ssqs[j], q); }
  }
  __syncthreads();
  if (tid < 16) { atomicAdd(&gsum[tid], ssum[tid]); atomicAdd(&gsqs[tid], ssqs[tid]); }
}

// Stage 2: h2 = elu(BN1(h1) @ W2^T + b2) in place; stats of h2.
__global__ void __launch_bounds__(256)
k_mlp1_s2(float* __restrict__ h, const float* __restrict__ w2,
          const float* __restrict__ b2,
          const float* __restrict__ scale, const float* __restrict__ shift,
          float* __restrict__ gsum, float* __restrict__ gsqs) {
  __shared__ float sw[256];
  __shared__ float ssum[16], ssqs[16];
  int tid = threadIdx.x;
  sw[tid] = w2[tid];
  if (tid < 16) { ssum[tid] = 0.f; ssqs[tid] = 0.f; }
  __syncthreads();
  long e = (long)blockIdx.x * 256 + tid;
  float a[16], o[16];
#pragma unroll
  for (int j = 0; j < 16; ++j) a[j] = h[e * 16 + j] * scale[j] + shift[j];
#pragma unroll
  for (int c = 0; c < 16; ++c) {
    float acc = b2[c];
#pragma unroll
    for (int j = 0; j < 16; ++j) acc += a[j] * sw[c * 16 + j];
    o[c] = elu1(acc);
    h[e * 16 + c] = o[c];
  }
#pragma unroll
  for (int c = 0; c < 16; ++c) {
    float s = wave_sum32(o[c]);
    float q = wave_sum32(o[c] * o[c]);
    if ((tid & 31) == 0) { atomicAdd(&ssum[c], s); atomicAdd(&ssqs[c], q); }
  }
  __syncthreads();
  if (tid < 16) { atomicAdd(&gsum[tid], ssum[tid]); atomicAdd(&gsqs[tid], ssqs[tid]); }
}

// ---------------------------------------------------------------------------
// mlp2 linear (WMMA): t1 = elu(rel48 @ wl^T + bl), [N,48]x[48,256], K pad 48->64.
// A/B fragments loaded as aligned v8h (16B) from f16 buffers.
__global__ void __launch_bounds__(256)
k_mlp2_linear(const _Float16* __restrict__ rel_h, const _Float16* __restrict__ wl_h,
              const float* __restrict__ bl, float* __restrict__ t,
              float* __restrict__ gsum, float* __restrict__ gsqs) {
  __shared__ float ssum[256], ssqs[256];
  int tid = threadIdx.x;
  ssum[tid] = 0.f; ssqs[tid] = 0.f;
  __syncthreads();
  int wv = tid >> 5, lane = tid & 31, m = lane & 15, hi = lane >> 4;
  long p0 = ((long)blockIdx.x * 8 + wv) * 16;

  const _Float16* arow = rel_h + (p0 + m) * 48;
  v8h zero8;
#pragma unroll
  for (int q = 0; q < 8; ++q) zero8[q] = (_Float16)0.f;
  v16h a0 = mk_frag(*(const v8h*)(arow + hi * 8), *(const v8h*)(arow + 16 + hi * 8));
  v16h a1 = mk_frag(*(const v8h*)(arow + 32 + hi * 8), zero8);   // K 48..63 pad

#pragma unroll 1
  for (int nt = 0; nt < 16; ++nt) {
    int nout = nt * 16 + m;
    const _Float16* brow = wl_h + nout * 48;
    v16h b0 = mk_frag(*(const v8h*)(brow + hi * 8), *(const v8h*)(brow + 16 + hi * 8));
    v16h b1 = mk_frag(*(const v8h*)(brow + 32 + hi * 8), zero8);
    float bias = bl[nout];
    v8f c;
#pragma unroll
    for (int v = 0; v < 8; ++v) c[v] = bias;
    c = wmma_f16f32(a0, b0, c);
    c = wmma_f16f32(a1, b1, c);
    float ps = 0.f, pq = 0.f;
#pragma unroll
    for (int v = 0; v < 8; ++v) {                      // D row = v + 8*hi
      float val = elu1(c[v]);
      t[(p0 + v + 8 * hi) * 256 + nout] = val;
      ps += val; pq += val * val;
    }
    atomicAdd(&ssum[nout], ps);
    atomicAdd(&ssqs[nout], pq);
  }
  __syncthreads();
  atomicAdd(&gsum[tid], ssum[tid]);
  atomicAdd(&gsqs[tid], ssqs[tid]);
}

// ---------------------------------------------------------------------------
// Grouped Conv1d(K,K^2,K,groups=K): out[g][j] = sum_t BN(in)[g][t]*cw[g][j][t]+cb.
__global__ void __launch_bounds__(256)
k_gconv(float* __restrict__ t, const float* __restrict__ cw,
        const float* __restrict__ cb,
        const float* __restrict__ scale, const float* __restrict__ shift,
        float* __restrict__ gsum, float* __restrict__ gsqs, int do_elu) {
  __shared__ float sin_[8][256];
  __shared__ float ssum[256], ssqs[256];
  int tid = threadIdx.x, wv = tid >> 5, lane = tid & 31;
  ssum[tid] = 0.f; ssqs[tid] = 0.f;
  long p = (long)blockIdx.x * 8 + wv;
#pragma unroll
  for (int q = 0; q < 8; ++q) {
    int idx = lane * 8 + q;
    sin_[wv][idx] = t[p * 256 + idx] * scale[idx] + shift[idx];
  }
  __syncthreads();
#pragma unroll
  for (int q = 0; q < 8; ++q) {
    int oi = lane * 8 + q;               // oi = g*16 + j
    int g = oi >> 4;
    float acc = cb[oi];
    const float* w  = cw + oi * 16;
    const float* in = &sin_[wv][g * 16];
#pragma unroll
    for (int tt = 0; tt < 16; ++tt) acc += in[tt] * w[tt];
    if (do_elu) acc = elu1(acc);
    t[p * 256 + oi] = acc;
    atomicAdd(&ssum[oi], acc);
    atomicAdd(&ssqs[oi], acc * acc);
  }
  __syncthreads();
  atomicAdd(&gsum[tid], ssum[tid]);
  atomicAdd(&gsqs[tid], ssqs[tid]);
}

// ---------------------------------------------------------------------------
// Core X-Conv (WMMA): per point, xt[80x16] = x_star[80x16] @ T'[16x16]^T
// (5 v_wmma, K padded 16->32), then depthwise conv -> y16[N,160] f16.
__global__ void __launch_bounds__(128)
k_xconv(const float* __restrict__ h, const _Float16* __restrict__ x_h,
        const int* __restrict__ col, const float* __restrict__ t,
        const float* __restrict__ sc2, const float* __restrict__ sh2,
        const float* __restrict__ scc, const float* __restrict__ shc,
        const float* __restrict__ dw, const float* __restrict__ db,
        _Float16* __restrict__ y16) {
  __shared__ float sxt[4][80][16];
  int tid = threadIdx.x, wv = tid >> 5, lane = tid & 31, m = lane & 15, hi = lane >> 4;
  long p = (long)blockIdx.x * 4 + wv;

  // B fragment: B^T row k = m of T' = BNc(traw).
  const float* trow = t + p * 256 + m * 16;
  v16h bf;
#pragma unroll
  for (int q = 0; q < 8; ++q) {
    int j = hi * 8 + q;
    bf[q]     = (_Float16)(trow[j] * scc[m * 16 + j] + shc[m * 16 + j]);
    bf[q + 8] = (_Float16)0.f;          // K 16..31 pad
  }

  const int* colrow = col + p * 16;
#pragma unroll 1
  for (int ct = 0; ct < 5; ++ct) {      // 5 tiles of 16 channels
    int c = ct * 16 + m;
    v16h af;
#pragma unroll
    for (int q = 0; q < 8; ++q) {
      int j = hi * 8 + q;               // neighbor index
      _Float16 val;
      if (c < 16) val = (_Float16)(h[((long)p * 16 + j) * 16 + c] * sc2[c] + sh2[c]);
      else        val = x_h[(long)colrow[j] * 64 + (c - 16)];
      af[q]     = val;
      af[q + 8] = (_Float16)0.f;
    }
    v8f acc;
#pragma unroll
    for (int v = 0; v < 8; ++v) acc[v] = 0.f;
    acc = wmma_f16f32(af, bf, acc);
#pragma unroll
    for (int v = 0; v < 8; ++v)
      sxt[wv][ct * 16 + v + 8 * hi][m] = acc[v];
  }
  __syncthreads();

  // Depthwise conv: y[i] = sum_t xt[i>>1][t]*dw[i][t] + db[i], i in 0..159.
#pragma unroll
  for (int r = 0; r < 5; ++r) {
    int i = lane + 32 * r;
    int c = i >> 1;
    float acc = db[i];
#pragma unroll
    for (int tt = 0; tt < 16; ++tt) acc += sxt[wv][c][tt] * dw[i * 16 + tt];
    y16[p * 160 + i] = (_Float16)acc;
  }
}

// ---------------------------------------------------------------------------
// Final linear (WMMA): out = y16[N,160] @ lw^T[160,128] + lb.
// One wave per 16 points: 8 N-tiles x 5 k-steps = 40 v_wmma.
__global__ void __launch_bounds__(256)
k_final(const _Float16* __restrict__ y16, const _Float16* __restrict__ lw_h,
        const float* __restrict__ lb, float* __restrict__ out) {
  int tid = threadIdx.x, wv = tid >> 5, lane = tid & 31, m = lane & 15, hi = lane >> 4;
  long p0 = ((long)blockIdx.x * 8 + wv) * 16;
  const _Float16* arow = y16 + (p0 + m) * 160;
  v16h a[5];
#pragma unroll
  for (int ks = 0; ks < 5; ++ks)
    a[ks] = mk_frag(*(const v8h*)(arow + ks * 32 + hi * 8),
                    *(const v8h*)(arow + ks * 32 + 16 + hi * 8));
#pragma unroll 1
  for (int nt = 0; nt < 8; ++nt) {
    int o = nt * 16 + m;
    const _Float16* brow = lw_h + o * 160;
    if (nt < 7) __builtin_prefetch(brow + 16 * 160, 0, 1);  // global_prefetch_b8
    float bias = lb[o];
    v8f c;
#pragma unroll
    for (int v = 0; v < 8; ++v) c[v] = bias;
#pragma unroll
    for (int ks = 0; ks < 5; ++ks) {
      v16h b = mk_frag(*(const v8h*)(brow + ks * 32 + hi * 8),
                       *(const v8h*)(brow + ks * 32 + 16 + hi * 8));
      c = wmma_f16f32(a[ks], b, c);
    }
#pragma unroll
    for (int v = 0; v < 8; ++v)
      out[(p0 + v + 8 * hi) * 128 + o] = c[v];
  }
}

// ---------------------------------------------------------------------------
extern "C" void kernel_launch(void* const* d_in, const int* in_sizes, int n_in,
                              void* d_out, int out_size, void* d_ws, size_t ws_size,
                              hipStream_t stream) {
  const float* x      = (const float*)d_in[0];
  const float* pos    = (const float*)d_in[1];
  const int*   col    = (const int*)d_in[2];
  const float* m1_w1  = (const float*)d_in[3];
  const float* m1_b1  = (const float*)d_in[4];
  const float* m1_g1  = (const float*)d_in[5];
  const float* m1_be1 = (const float*)d_in[6];
  const float* m1_w2  = (const float*)d_in[7];
  const float* m1_b2  = (const float*)d_in[8];
  const float* m1_g2  = (const float*)d_in[9];
  const float* m1_be2 = (const float*)d_in[10];
  const float* m2_wl  = (const float*)d_in[11];
  const float* m2_bl  = (const float*)d_in[12];
  const float* m2_ga  = (const float*)d_in[13];
  const float* m2_ba  = (const float*)d_in[14];
  const float* m2_cw1 = (const float*)d_in[15];
  const float* m2_cb1 = (const float*)d_in[16];
  const float* m2_gb  = (const float*)d_in[17];
  const float* m2_bb  = (const float*)d_in[18];
  const float* m2_cw2 = (const float*)d_in[19];
  const float* m2_cb2 = (const float*)d_in[20];
  const float* m2_gc  = (const float*)d_in[21];
  const float* m2_bc  = (const float*)d_in[22];
  const float* dw     = (const float*)d_in[23];
  const float* db     = (const float*)d_in[24];
  const float* lw     = (const float*)d_in[25];
  const float* lb     = (const float*)d_in[26];
  float* out = (float*)d_out;

  // Workspace carve-up (all f16 regions 16B-aligned).
  float*    t     = (float*)d_ws;                         // [N,256]
  float*    h     = t + (size_t)XN * 256;                 // [NE,16]
  _Float16* rel_h = (_Float16*)(h + (size_t)XNE * 16);    // [NE,3]
  _Float16* y16   = rel_h + (size_t)XNE * 3;              // [N,160]
  _Float16* x_h   = y16 + (size_t)XN * 160;               // [N,64]
  _Float16* wl_h  = x_h + (size_t)XN * 64;                // [256,48]
  _Float16* lw_h  = wl_h + 256 * 48;                      // [128,160]
  float*    stats = (float*)(lw_h + 128 * 160);           // 3200 f32
  float *s1sum = stats +    0, *s1sqs = stats +   16, *s1sc = stats +   32, *s1sh = stats +   48;
  float *s2sum = stats +   64, *s2sqs = stats +   80, *s2sc = stats +   96, *s2sh = stats +  112;
  float *asum  = stats +  128, *asqs  = stats +  384, *asc  = stats +  640, *ash  = stats +  896;
  float *bsum  = stats + 1152, *bsqs  = stats + 1408, *bsc  = stats + 1664, *bsh  = stats + 1920;
  float *csum  = stats + 2176, *csqs  = stats + 2432, *csc  = stats + 2688, *csh  = stats + 2944;

  const float invNE = 1.f / (float)XNE;
  const float invN  = 1.f / (float)XN;

  k_zero<<<(3200 + 255) / 256, 256, 0, stream>>>(stats, 3200);
  // One-time f16 conversions (weights tiny, x = 8MB, all L2-resident after).
  k_tohalf<<<(256 * 48 + 255) / 256, 256, 0, stream>>>(m2_wl, wl_h, 256 * 48);
  k_tohalf<<<(128 * 160 + 255) / 256, 256, 0, stream>>>(lw, lw_h, 128 * 160);
  k_tohalf<<<(XN * 64 + 255) / 256, 256, 0, stream>>>(x, x_h, XN * 64);

  // mlp1 (per-edge)
  k_mlp1_s1<<<XNE / 256, 256, 0, stream>>>(pos, col, m1_w1, m1_b1, rel_h, h, s1sum, s1sqs);
  k_finalize_bn<<<1, 256, 0, stream>>>(s1sum, s1sqs, m1_g1, m1_be1, s1sc, s1sh, 16, invNE);
  k_mlp1_s2<<<XNE / 256, 256, 0, stream>>>(h, m1_w2, m1_b2, s1sc, s1sh, s2sum, s2sqs);
  k_finalize_bn<<<1, 256, 0, stream>>>(s2sum, s2sqs, m1_g2, m1_be2, s2sc, s2sh, 16, invNE);

  // mlp2 (per-point X-transform)
  k_mlp2_linear<<<XN / 128, 256, 0, stream>>>(rel_h, wl_h, m2_bl, t, asum, asqs);
  k_finalize_bn<<<1, 256, 0, stream>>>(asum, asqs, m2_ga, m2_ba, asc, ash, 256, invN);
  k_gconv<<<XN / 8, 256, 0, stream>>>(t, m2_cw1, m2_cb1, asc, ash, bsum, bsqs, 1);
  k_finalize_bn<<<1, 256, 0, stream>>>(bsum, bsqs, m2_gb, m2_bb, bsc, bsh, 256, invN);
  k_gconv<<<XN / 8, 256, 0, stream>>>(t, m2_cw2, m2_cb2, bsc, bsh, csum, csqs, 0);
  k_finalize_bn<<<1, 256, 0, stream>>>(csum, csqs, m2_gc, m2_bc, csc, csh, 256, invN);

  // core transform + depthwise conv, then final linear
  k_xconv<<<XN / 4, 128, 0, stream>>>(h, x_h, col, t, s2sc, s2sh, csc, csh, dw, db, y16);
  k_final<<<XN / 128, 256, 0, stream>>>(y16, lw_h, lb, out);
}
